// RealisticDelays_37495064494225
// MI455X (gfx1250) — compile-verified, matrix operand order checked by reading, not run on hardware
//
#include <hip/hip_runtime.h>

// ---------------------------------------------------------------------------
// RealisticDelays: out[b][t] = sum_s buf[(cur - floor(delays[s][t]/DT)) % 21, s]
// buf = spike_buffer with row `cur` replaced by input_spikes[0].
//
// Memory-bound: delays = 256 MB streamed once -> ~11 us @ 23.3 TB/s.
// Spikes are {0,1}: fold the 21-deep ring column per source into a 21-bit
// mask so the per-element work is  acc += (mask >> steps) & 1  (5 VALU ops).
// Cross-chunk reduction (32 x 8192 column sum) done with
// v_wmma_f32_16x16x32_f16 using an all-ones A matrix (K-order invariant).
// ---------------------------------------------------------------------------

#define S_SIZE   8192
#define T_SIZE   8192
#define D_STEPS  21
#define BATCH    4

#define NCHUNK       32                      // S split into 32 chunks of 256
#define CHUNK        (S_SIZE / NCHUNK)       // 256 sources per chunk
#define BLOCK        256                     // threads per block (8 waves)
#define T_PER_THREAD 4                       // float4 delay loads
#define T_PER_BLOCK  (BLOCK * T_PER_THREAD)  // 1024 targets per block
#define T_TILES      (T_SIZE / T_PER_BLOCK)  // 8

typedef __attribute__((ext_vector_type(16))) _Float16 v16h;
typedef __attribute__((ext_vector_type(8)))  float    v8f;

// ---------------------------------------------------------------------------
// Kernel A: per-source bitmask.  bit k of masks[s] = buf[(cur-k) mod 21, s]
// (k = 0 hits row cur, i.e. the freshly written input_spikes[0] frame).
// Shifts by steps in [21,31] naturally return 0 => `valid` is free.
// ---------------------------------------------------------------------------
__global__ void rd_build_masks(const float* __restrict__ input_spikes,
                               const float* __restrict__ spike_buffer,
                               const int*   __restrict__ buffer_index,
                               unsigned*    __restrict__ masks) {
  int s = blockIdx.x * blockDim.x + threadIdx.x;
  if (s >= S_SIZE) return;
  int cur = buffer_index[0] % D_STEPS;
  if (cur < 0) cur += D_STEPS;

  unsigned m = 0u;
  if (input_spikes[s] != 0.0f) m |= 1u;          // k = 0 -> current frame
  int d = cur;
  #pragma unroll
  for (int k = 1; k < D_STEPS; ++k) {
    d -= 1; if (d < 0) d += D_STEPS;             // d = (cur - k) mod 21
    if (spike_buffer[(size_t)d * S_SIZE + s] != 0.0f) m |= (1u << k);
  }
  masks[s] = m;
}

// ---------------------------------------------------------------------------
// Kernel B: the streaming hot loop. grid = (T_TILES, NCHUNK).
// Each thread owns 4 target columns, loops over its 256-source chunk:
// one coalesced float4 delay load + one wave-uniform LDS mask broadcast,
// then 5 VALU ops per element. Writes integer partial sums (<= 256, f16-exact).
// ---------------------------------------------------------------------------
__global__ void rd_gather_partial(const float*    __restrict__ delays,
                                  const unsigned* __restrict__ masks,
                                  float*          __restrict__ partials) {
  __shared__ unsigned lds_mask[CHUNK];

  const int chunk = blockIdx.y;
  const int s0    = chunk * CHUNK;
  lds_mask[threadIdx.x] = masks[s0 + threadIdx.x];   // CHUNK == BLOCK
  __syncthreads();

  const int t0 = blockIdx.x * T_PER_BLOCK + threadIdx.x * T_PER_THREAD;
  const float* p = delays + (size_t)s0 * T_SIZE + t0;

  unsigned a0 = 0u, a1 = 0u, a2 = 0u, a3 = 0u;
  #pragma unroll 4
  for (int sl = 0; sl < CHUNK; ++sl) {
    float4 dl = *(const float4*)p;
    p += T_SIZE;
    unsigned m = lds_mask[sl];                       // wave-uniform broadcast
    // steps = trunc(delay * 1000) ; delay > 0 so trunc == floor
    a0 += (m >> (int)(dl.x * 1000.0f)) & 1u;
    a1 += (m >> (int)(dl.y * 1000.0f)) & 1u;
    a2 += (m >> (int)(dl.z * 1000.0f)) & 1u;
    a3 += (m >> (int)(dl.w * 1000.0f)) & 1u;
  }

  float4 r = make_float4((float)a0, (float)a1, (float)a2, (float)a3);
  *(float4*)(partials + (size_t)chunk * T_SIZE + t0) = r;
}

// ---------------------------------------------------------------------------
// Kernel C: column-sum of partials[32][T] on the matrix pipe.
// D = ones(16x32) x B(32x16) + 0  =>  every row of D = per-column sum over the
// 32 chunks. A == ones makes the result invariant to the B K-ordering; lane L
// holds column n = L & 15 per the ISA B layout. c[0] of every lane is the full
// sum for its column. Broadcast to all 4 batch rows.
// ---------------------------------------------------------------------------
__global__ void rd_reduce_wmma(const float* __restrict__ partials,
                               float*       __restrict__ out) {
  const int lane   = threadIdx.x & 31;
  const int wave   = threadIdx.x >> 5;
  const int t_base = blockIdx.x * 128 + wave * 16;   // 8 waves * 16 targets
  const int n      = lane & 15;
  const int kb     = (lane >> 4) * 16;               // lanes 0-15: K 0-15; 16-31: K 16-31
  const int t      = t_base + n;

  v16h ones;
  #pragma unroll
  for (int i = 0; i < 16; ++i) ones[i] = (_Float16)1.0f;

  v16h b;
  #pragma unroll
  for (int i = 0; i < 16; ++i) {
    float v = partials[(size_t)(kb + i) * T_SIZE + t];  // counts <= 256: f16-exact
    b[i] = (_Float16)v;
  }

  v8f c = {};
  c = __builtin_amdgcn_wmma_f32_16x16x32_f16(
      /*neg_a=*/false, ones, /*neg_b=*/false, b,
      /*c_mod=*/(short)0, c, /*reuse_a=*/false, /*reuse_b=*/false);

  float r = c[0];   // (M=0,N=n) in lanes 0-15, (M=8,N=n) in lanes 16-31: same sum
  #pragma unroll
  for (int bi = 0; bi < BATCH; ++bi)
    out[(size_t)bi * T_SIZE + t] = r;
}

// ---------------------------------------------------------------------------
extern "C" void kernel_launch(void* const* d_in, const int* in_sizes, int n_in,
                              void* d_out, int out_size, void* d_ws, size_t ws_size,
                              hipStream_t stream) {
  (void)in_sizes; (void)n_in; (void)out_size; (void)ws_size;

  const float* input_spikes = (const float*)d_in[0];   // [4, 8192]
  const float* delays       = (const float*)d_in[1];   // [8192, 8192]
  const float* spike_buffer = (const float*)d_in[2];   // [21, 8192]
  const int*   buffer_index = (const int*)  d_in[3];   // scalar
  float* out = (float*)d_out;                          // [4, 8192]

  unsigned* masks    = (unsigned*)d_ws;                            // 32 KB
  float*    partials = (float*)((char*)d_ws + S_SIZE * sizeof(unsigned)); // 1 MB

  rd_build_masks<<<S_SIZE / 256, 256, 0, stream>>>(
      input_spikes, spike_buffer, buffer_index, masks);

  dim3 gridB(T_TILES, NCHUNK);
  rd_gather_partial<<<gridB, BLOCK, 0, stream>>>(delays, masks, partials);

  rd_reduce_wmma<<<T_SIZE / 128, 256, 0, stream>>>(partials, out);
}
